// GMHCN_51866025067049
// MI455X (gfx1250) — matrix operand order repaired and checked by reading, not compiled, not run on hardware
//
#include <hip/hip_runtime.h>
#include <math.h>

#define NN 4096
#define EE 16384
#define HH 6

typedef __attribute__((ext_vector_type(16))) _Float16 v16h;
typedef __attribute__((ext_vector_type(8)))  float    v8f;

__device__ __forceinline__ v16h cvt16(float4 a, float4 b, float4 c, float4 d) {
  v16h r;
  r[0]  = (_Float16)a.x; r[1]  = (_Float16)a.y; r[2]  = (_Float16)a.z; r[3]  = (_Float16)a.w;
  r[4]  = (_Float16)b.x; r[5]  = (_Float16)b.y; r[6]  = (_Float16)b.z; r[7]  = (_Float16)b.w;
  r[8]  = (_Float16)c.x; r[9]  = (_Float16)c.y; r[10] = (_Float16)c.z; r[11] = (_Float16)c.w;
  r[12] = (_Float16)d.x; r[13] = (_Float16)d.y; r[14] = (_Float16)d.z; r[15] = (_Float16)d.w;
  return r;
}

// ---------------------------------------------------------------------------
// Weight transpose: W (K x N row-major) -> Wt (N x K row-major).
// One-shot, coalesced on the read side; cost negligible vs the GEMM.
// ---------------------------------------------------------------------------
__global__ void transpose_w(const float* __restrict__ W, float* __restrict__ Wt,
                            int K, int N) {
  int idx = blockIdx.x * blockDim.x + threadIdx.x;   // = k*N + n
  if (idx >= K * N) return;
  int k = idx / N, n = idx % N;
  Wt[(size_t)n * K + k] = W[idx];
}

// ---------------------------------------------------------------------------
// WMMA GEMM:  C[M,N] = A[M,K] @ B[K,N] (+ bias[N]),  B passed TRANSPOSED (NxK).
// wave32; each wave computes a 64(M) x 32(N) tile => 8 accumulators; A frags
// reused 2x, B frags reused 4x. Block = 8 waves => 64 x 256 of C.
// fp32 in, f16 WMMA, f32 accumulate.
// REQUIREMENTS (guaranteed by all call sites):
//   - M % 64 == 0 (M == 4096)
//   - K % 4 == 0 (so rows of A and Bt are 16-byte aligned)
// All hot-loop loads are unconditional b128 with immediate offsets; columns
// past N are clamped (their results are computed but never stored).
// ---------------------------------------------------------------------------
__global__ __launch_bounds__(256)
void wmma_gemm(const float* __restrict__ A, const float* __restrict__ Bt,
               const float* __restrict__ bias, float* __restrict__ C,
               int M, int N, int K) {
  const int lane = threadIdx.x & 31;
  const int wave = threadIdx.x >> 5;
  const int n0   = blockIdx.x * 256 + wave * 32;   // wave-uniform
  const int m0   = blockIdx.y * 64;                // wave-uniform
  if (n0 >= N) return;                             // whole wave exits together

  const int  sub = lane & 15;     // row (A frag) / col (B,C frags)
  const int  grp = lane >> 4;     // 0 or 1

  const int  col0  = n0 + sub;
  const int  col1  = n0 + 16 + sub;
  const bool ok0   = col0 < N;
  const bool ok1   = col1 < N;
  const int  colc0 = ok0 ? col0 : (N - 1);
  const int  colc1 = ok1 ? col1 : (N - 1);

  const float* Bc0 = Bt + (size_t)colc0 * K;       // contiguous K floats
  const float* Bc1 = Bt + (size_t)colc1 * K;
  const float* Ar[4];
#pragma unroll
  for (int t = 0; t < 4; ++t) Ar[t] = A + (size_t)(m0 + t * 16 + sub) * K;

  v8f acc[4][2] = {};

  const int Kfull = K & ~31;
  int k0 = 0;
  // ---------------- main loop: full 32-wide K tiles, no masking ------------
  for (; k0 < Kfull; k0 += 32) {
    // B fragments: bf[i] = Bt[col][k0 + 16*grp + i]  (16 contiguous floats)
    const float4* bp0 = (const float4*)(Bc0 + k0 + grp * 16);
    v16h bf0 = cvt16(bp0[0], bp0[1], bp0[2], bp0[3]);
    const float4* bp1 = (const float4*)(Bc1 + k0 + grp * 16);
    v16h bf1 = cvt16(bp1[0], bp1[1], bp1[2], bp1[3]);

#pragma unroll
    for (int t = 0; t < 4; ++t) {
      // A fragment: af[i] = A[row, k0 + 16*grp + i]  (16 contiguous floats)
      const float4* ap = (const float4*)(Ar[t] + k0 + grp * 16);
      v16h af = cvt16(ap[0], ap[1], ap[2], ap[3]);
      acc[t][0] = __builtin_amdgcn_wmma_f32_16x16x32_f16(
          false, af, false, bf0, (short)0, acc[t][0], false, false);
      acc[t][1] = __builtin_amdgcn_wmma_f32_16x16x32_f16(
          false, af, false, bf1, (short)0, acc[t][1], false, false);
    }
  }
  // ---------------- epilogue: partial K tile, clamped loads + selects ------
  if (k0 < K) {
    v16h bf0, bf1;
#pragma unroll
    for (int i = 0; i < 16; ++i) {
      int kk = k0 + grp * 16 + i;
      int kc = kk < K ? kk : (K - 1);
      float x0 = Bc0[kc];
      float x1 = Bc1[kc];
      bf0[i] = (_Float16)(kk < K ? x0 : 0.0f);
      bf1[i] = (_Float16)(kk < K ? x1 : 0.0f);
    }
#pragma unroll
    for (int t = 0; t < 4; ++t) {
      v16h af;
#pragma unroll
      for (int i = 0; i < 16; ++i) {
        int kk = k0 + grp * 16 + i;
        int kc = kk < K ? kk : (K - 1);
        float x = Ar[t][kc];
        af[i] = (_Float16)(kk < K ? x : 0.0f);
      }
      acc[t][0] = __builtin_amdgcn_wmma_f32_16x16x32_f16(
          false, af, false, bf0, (short)0, acc[t][0], false, false);
      acc[t][1] = __builtin_amdgcn_wmma_f32_16x16x32_f16(
          false, af, false, bf1, (short)0, acc[t][1], false, false);
    }
  }

  // ---------------- store (C layout: row = v + 8*grp within tile) ----------
  if (ok0) {
    float bv = bias ? bias[col0] : 0.0f;
#pragma unroll
    for (int t = 0; t < 4; ++t)
#pragma unroll
      for (int v = 0; v < 8; ++v)
        C[(size_t)(m0 + t * 16 + v + 8 * grp) * N + col0] = acc[t][0][v] + bv;
  }
  if (ok1) {
    float bv = bias ? bias[col1] : 0.0f;
#pragma unroll
    for (int t = 0; t < 4; ++t)
#pragma unroll
      for (int v = 0; v < 8; ++v)
        C[(size_t)(m0 + t * 16 + v + 8 * grp) * N + col1] = acc[t][1][v] + bv;
  }
}

// ---------------------------------------------------------------------------
// Elementwise / graph kernels
// ---------------------------------------------------------------------------
__global__ void fill_f(float* p, float v, size_t n) {
  size_t i = (size_t)blockIdx.x * blockDim.x + threadIdx.x;
  size_t st = (size_t)gridDim.x * blockDim.x;
  for (; i < n; i += st) p[i] = v;
}

__global__ void degree_count(const int* __restrict__ src, const int* __restrict__ dst,
                             float* degO, float* degI) {
  int e = blockIdx.x * blockDim.x + threadIdx.x;
  if (e < EE) {
    atomicAdd(&degO[src[e]], 1.0f);
    atomicAdd(&degI[dst[e]], 1.0f);
  }
}

__global__ void to_invsqrt(float* d, int n) {
  int i = blockIdx.x * blockDim.x + threadIdx.x;
  if (i < n) d[i] = 1.0f / sqrtf(fmaxf(d[i], 1.0f));
}

__global__ void scale_rows(const float* __restrict__ in, const float* __restrict__ s,
                           float* __restrict__ out, int F) {
  int n = blockIdx.y;
  int f = blockIdx.x * blockDim.x + threadIdx.x;
  if (f < F) out[(size_t)n * F + f] = in[(size_t)n * F + f] * s[n];
}

__global__ void gcn_scatter(const int* __restrict__ src, const int* __restrict__ dst,
                            const float* __restrict__ h, float* __restrict__ agg, int F) {
  int e = blockIdx.y;
  int s = src[e], d = dst[e];
  int f = blockIdx.x * blockDim.x + threadIdx.x;
  if (f < F) atomicAdd(&agg[(size_t)d * F + f], h[(size_t)s * F + f]);
}

// ---------------------------------------------------------------------------
// GAT attention kernels
// ---------------------------------------------------------------------------
__global__ void gat_scores(const float* __restrict__ h, const float* __restrict__ al,
                           const float* __restrict__ ar, float* el, float* er, int F) {
  int i = blockIdx.x * blockDim.x + threadIdx.x;
  if (i >= NN * HH) return;
  int n = i / HH, hd = i % HH;
  const float* hp  = h + (size_t)n * HH * F + (size_t)hd * F;
  const float* alp = al + hd * F;
  const float* arp = ar + hd * F;
  float sl = 0.0f, sr = 0.0f;
  for (int f = 0; f < F; ++f) {
    float v = hp[f];
    sl += v * alp[f];
    sr += v * arp[f];
  }
  el[i] = sl;
  er[i] = sr;
}

__device__ __forceinline__ void atomicMaxF(float* addr, float val) {
  // monotone encodings: ints for positives, reversed unsigned for negatives
  if (val >= 0.0f) atomicMax((int*)addr, __float_as_int(val));
  else             atomicMin((unsigned int*)addr, __float_as_uint(val));
}

__global__ void gat_edge_max(const int* __restrict__ src, const int* __restrict__ dst,
                             const float* __restrict__ el, const float* __restrict__ er,
                             float* emax) {
  int i = blockIdx.x * blockDim.x + threadIdx.x;
  if (i >= EE * HH) return;
  int e = i / HH, hd = i % HH;
  float v = el[src[e] * HH + hd] + er[dst[e] * HH + hd];
  v = v > 0.0f ? v : 0.2f * v;              // leaky_relu(0.2)
  atomicMaxF(&emax[dst[e] * HH + hd], v);
}

__global__ void fix_emax(float* emax, int n) {
  int i = blockIdx.x * blockDim.x + threadIdx.x;
  if (i < n) {
    float v = emax[i];
    if (!(v >= -3.4e38f)) emax[i] = 0.0f;   // -inf / NaN -> 0
  }
}

__global__ void gat_edge_exp(const int* __restrict__ src, const int* __restrict__ dst,
                             const float* __restrict__ el, const float* __restrict__ er,
                             const float* __restrict__ emax, float* den, float* ee) {
  int i = blockIdx.x * blockDim.x + threadIdx.x;
  if (i >= EE * HH) return;
  int e = i / HH, hd = i % HH;
  float v = el[src[e] * HH + hd] + er[dst[e] * HH + hd];
  v = v > 0.0f ? v : 0.2f * v;
  float x = expf(v - emax[dst[e] * HH + hd]);
  ee[i] = x;
  atomicAdd(&den[dst[e] * HH + hd], x);
}

__global__ void gat_scatter(const int* __restrict__ src, const int* __restrict__ dst,
                            const float* __restrict__ h, const float* __restrict__ ee,
                            float* __restrict__ num, int F) {
  int e = blockIdx.y;
  int s = src[e], d = dst[e];
  int HF = HH * F;
  int j = blockIdx.x * blockDim.x + threadIdx.x;
  if (j < HF) {
    int hd = j / F;
    atomicAdd(&num[(size_t)d * HF + j], h[(size_t)s * HF + j] * ee[e * HH + hd]);
  }
}

__global__ void gat_finalize(float* __restrict__ out, const float* __restrict__ den,
                             const float* __restrict__ b, int F) {
  int HF = HH * F;
  size_t i = (size_t)blockIdx.x * blockDim.x + threadIdx.x;
  if (i >= (size_t)NN * HF) return;
  int n  = (int)(i / HF);
  int j  = (int)(i % HF);
  int hd = j / F;
  float dn = den[n * HH + hd];
  out[i] = out[i] / (dn > 0.0f ? dn : 1.0f) + b[j];
}

// ---------------------------------------------------------------------------
// Conv / pool / deconv kernels
// ---------------------------------------------------------------------------
__global__ void conv3x3(const float* __restrict__ in, const float* __restrict__ W,
                        const float* __restrict__ b, float* __restrict__ out,
                        int Cin, int Cout, int Hin, int Win, int doRelu, int total) {
  int idx = blockIdx.x * blockDim.x + threadIdx.x;
  if (idx >= total) return;
  int Ho = Hin - 2, Wo = Win - 2;
  int x = idx % Wo; int t = idx / Wo;
  int y = t % Ho;  t /= Ho;
  int co = t % Cout; int n = t / Cout;
  const float* ip = in + (size_t)n * Cin * Hin * Win;
  const float* wp = W + (size_t)co * Cin * 9;     // OIHW
  float s = b[co];
  for (int ci = 0; ci < Cin; ++ci) {
    const float* ipc = ip + (size_t)ci * Hin * Win + y * Win + x;
    const float* wpc = wp + ci * 9;
#pragma unroll
    for (int ky = 0; ky < 3; ++ky)
#pragma unroll
      for (int kx = 0; kx < 3; ++kx)
        s += ipc[ky * Win + kx] * wpc[ky * 3 + kx];
  }
  if (doRelu) s = fmaxf(s, 0.0f);
  out[idx] = s;
}

__global__ void maxpool_k(const float* __restrict__ in, float* __restrict__ out,
                          int C, int Hin, int Win, int k, int st, int total) {
  int idx = blockIdx.x * blockDim.x + threadIdx.x;
  if (idx >= total) return;
  int Ho = (Hin - k) / st + 1, Wo = (Win - k) / st + 1;
  int x = idx % Wo; int t = idx / Wo;
  int y = t % Ho;  t /= Ho;
  int c = t % C; int n = t / C;
  const float* ip = in + ((size_t)n * C + c) * Hin * Win;
  float m = -INFINITY;
  for (int ky = 0; ky < k; ++ky)
    for (int kx = 0; kx < k; ++kx)
      m = fmaxf(m, ip[(y * st + ky) * Win + (x * st + kx)]);
  out[idx] = m;
}

// ConvTranspose2d stride 1, k=3: out[y,x] = b + sum W[ci,co,ay,ax]*in[ci,y-ay,x-ax]
__global__ void convt3x3(const float* __restrict__ in, const float* __restrict__ W,
                         const float* __restrict__ b, float* __restrict__ out,
                         int Cin, int Cout, int Hin, int Win, int act, int total) {
  int idx = blockIdx.x * blockDim.x + threadIdx.x;
  if (idx >= total) return;
  int Ho = Hin + 2, Wo = Win + 2;
  int x = idx % Wo; int t = idx / Wo;
  int y = t % Ho;  t /= Ho;
  int co = t % Cout; int n = t / Cout;
  float s = b[co];
  for (int ci = 0; ci < Cin; ++ci) {
    const float* ipc = in + ((size_t)n * Cin + ci) * Hin * Win;
    const float* wpc = W + ((size_t)ci * Cout + co) * 9;   // (Cin,Cout,3,3)
#pragma unroll
    for (int ay = 0; ay < 3; ++ay) {
      int yy = y - ay;
      if (yy < 0 || yy >= Hin) continue;
#pragma unroll
      for (int ax = 0; ax < 3; ++ax) {
        int xx = x - ax;
        if (xx < 0 || xx >= Win) continue;
        s += ipc[yy * Win + xx] * wpc[ay * 3 + ax];
      }
    }
  }
  if (act == 1) s = fmaxf(s, 0.0f);
  else if (act == 2) s = tanhf(s);
  out[idx] = s;
}

// ---------------------------------------------------------------------------
// Host orchestration
// ---------------------------------------------------------------------------
enum {
  P_gcn1_W = 0, P_gcn1_b,
  P_gat1_W, P_gat1_al, P_gat1_ar, P_gat1_b,
  P_gcn2_W, P_gcn2_b,
  P_gat2_W, P_gat2_al, P_gat2_ar, P_gat2_b,
  P_gcn3_W, P_gcn3_b,
  P_gat3_W, P_gat3_al, P_gat3_ar, P_gat3_b,
  P_enc1_W, P_enc1_b, P_enc2_W, P_enc2_b,
  P_ma_W, P_ma_al, P_ma_ar, P_ma_b,
  P_gcn4_W, P_gcn4_b,
  P_gat4_W, P_gat4_al, P_gat4_ar, P_gat4_b,
  P_gcn5_W, P_gcn5_b,
  P_gat5_W, P_gat5_al, P_gat5_ar, P_gat5_b,
  P_gcn6_W, P_gcn6_b,
  P_gat6_W, P_gat6_al, P_gat6_ar, P_gat6_b,
  P_dec1_W, P_dec1_b, P_dec2_W, P_dec2_b, P_dec3_W, P_dec3_b,
  P_dense_W, P_dense_b
};

extern "C" void kernel_launch(void* const* d_in, const int* in_sizes, int n_in,
                              void* d_out, int out_size, void* d_ws, size_t ws_size,
                              hipStream_t stream) {
  const float* features = (const float*)d_in[0];
  const int*   src      = (const int*)d_in[n_in - 2];
  const int*   dst      = (const int*)d_in[n_in - 1];
  auto P = [&](int i) { return (const float*)d_in[1 + i]; };

  // ---- workspace layout (floats) ----
  const size_t NB = (size_t)NN * 9600;
  float* BIG0  = (float*)d_ws;
  float* BIG1  = BIG0 + NB;
  float* BIG2  = BIG1 + NB;
  float* dinvO = BIG2 + NB;                 // N   (also deg_out during count)
  float* dinvI = dinvO + NN;                // N
  float* el    = dinvI + NN;                // N*H
  float* er    = el + (size_t)NN * HH;
  float* emax  = er + (size_t)NN * HH;
  float* den   = emax + (size_t)NN * HH;
  float* ee    = den + (size_t)NN * HH;     // E*H
  float* WT    = ee + (size_t)EE * HH;      // transposed weights, N*1200 floats
  size_t need  = (size_t)(WT + (size_t)NN * 1200 - BIG0) * sizeof(float);
  if (ws_size < need) return;               // workspace too small: do nothing

  auto fill = [&](float* p, float v, size_t n) {
    size_t blocks = (n + 255) / 256;
    if (blocks > 65535) blocks = 65535;
    fill_f<<<(unsigned)blocks, 256, 0, stream>>>(p, v, n);
  };
  auto gemm = [&](const float* A, const float* B, const float* bias, float* C,
                  int M, int N, int K) {
    int tot = K * N;
    transpose_w<<<(tot + 255) / 256, 256, 0, stream>>>(B, WT, K, N);
    dim3 grid((N + 255) / 256, (M + 63) / 64);
    wmma_gemm<<<grid, 256, 0, stream>>>(A, WT, bias, C, M, N, K);
  };
  auto run_gcn = [&](const float* x, int K, int Nc, const float* W, const float* b,
                     float* out, float* tmp, float* agg) {
    dim3 gn((K + 255) / 256, NN);
    scale_rows<<<gn, 256, 0, stream>>>(x, dinvO, tmp, K);
    fill(agg, 0.0f, (size_t)NN * K);
    dim3 ge((K + 255) / 256, EE);
    gcn_scatter<<<ge, 256, 0, stream>>>(src, dst, tmp, agg, K);
    scale_rows<<<gn, 256, 0, stream>>>(agg, dinvI, agg, K);
    gemm(agg, W, b, out, NN, Nc, K);
  };
  auto run_gat = [&](const float* x, int K, int F, const float* W, const float* al_,
                     const float* ar_, const float* b_, float* h, float* out) {
    int HF = HH * F;
    gemm(x, W, nullptr, h, NN, HF, K);
    gat_scores<<<(NN * HH + 255) / 256, 256, 0, stream>>>(h, al_, ar_, el, er, F);
    fill(emax, -INFINITY, (size_t)NN * HH);
    gat_edge_max<<<(EE * HH + 255) / 256, 256, 0, stream>>>(src, dst, el, er, emax);
    fix_emax<<<(NN * HH + 255) / 256, 256, 0, stream>>>(emax, NN * HH);
    fill(den, 0.0f, (size_t)NN * HH);
    gat_edge_exp<<<(EE * HH + 255) / 256, 256, 0, stream>>>(src, dst, el, er, emax, den, ee);
    fill(out, 0.0f, (size_t)NN * HF);
    dim3 gs((HF + 255) / 256, EE);
    gat_scatter<<<gs, 256, 0, stream>>>(src, dst, h, ee, out, F);
    size_t tot = (size_t)NN * HF;
    gat_finalize<<<(unsigned)((tot + 255) / 256), 256, 0, stream>>>(out, den, b_, F);
  };

  // ---- degree normalizers (in-place -> D^{-1/2}) ----
  fill(dinvO, 0.0f, 2 * NN);
  degree_count<<<(EE + 255) / 256, 256, 0, stream>>>(src, dst, dinvO, dinvI);
  to_invsqrt<<<(2 * NN + 255) / 256, 256, 0, stream>>>(dinvO, 2 * NN);

  float* g2a = BIG2;                        // gcn tmp   (<= N*1536)
  float* g2b = BIG2 + (size_t)NN * 4800;    // gcn agg   (<= N*1536)

  // ---- graph stack (activations ping-pong BIG0 <-> BIG1, scratch BIG2) ----
  run_gcn(features, 64, 36,   P(P_gcn1_W), P(P_gcn1_b), BIG0, g2a, g2b);
  run_gat(BIG0, 36, 12,   P(P_gat1_W), P(P_gat1_al), P(P_gat1_ar), P(P_gat1_b), BIG2, BIG1);
  run_gcn(BIG1, 72, 128,  P(P_gcn2_W), P(P_gcn2_b), BIG0, g2a, g2b);
  run_gat(BIG0, 128, 256, P(P_gat2_W), P(P_gat2_al), P(P_gat2_ar), P(P_gat2_b), BIG2, BIG1);
  run_gcn(BIG1, 1536, 512, P(P_gcn3_W), P(P_gcn3_b), BIG0, g2a, g2b);
  run_gat(BIG0, 512, 1600, P(P_gat3_W), P(P_gat3_al), P(P_gat3_ar), P(P_gat3_b), BIG2, BIG1);

  // ---- CNN encoder: BIG1 = (N,96,10,10) -> (N,32) in BIG0 ----
  {
    float* t0 = BIG2;                          // (N,64,8,8)
    float* t1 = BIG2 + (size_t)NN * 4096;      // (N,64,4,4)
    float* t2 = t1 + (size_t)NN * 1024;        // (N,32,2,2)
    int tot;
    tot = NN * 64 * 8 * 8;
    conv3x3<<<(tot + 255) / 256, 256, 0, stream>>>(BIG1, P(P_enc1_W), P(P_enc1_b), t0,
                                                   96, 64, 10, 10, 1, tot);
    tot = NN * 64 * 4 * 4;
    maxpool_k<<<(tot + 255) / 256, 256, 0, stream>>>(t0, t1, 64, 8, 8, 2, 2, tot);
    tot = NN * 32 * 2 * 2;
    conv3x3<<<(tot + 255) / 256, 256, 0, stream>>>(t1, P(P_enc2_W), P(P_enc2_b), t2,
                                                   64, 32, 4, 4, 1, tot);
    tot = NN * 32;
    maxpool_k<<<(tot + 255) / 256, 256, 0, stream>>>(t2, BIG0, 32, 2, 2, 2, 1, tot);
  }

  // ---- remaining graph stack ----
  run_gat(BIG0, 32, 16,  P(P_ma_W),  P(P_ma_al),  P(P_ma_ar),  P(P_ma_b),  BIG2, BIG1);
  run_gcn(BIG1, 96, 100, P(P_gcn4_W), P(P_gcn4_b), BIG0, g2a, g2b);
  run_gat(BIG0, 100, 50, P(P_gat4_W), P(P_gat4_al), P(P_gat4_ar), P(P_gat4_b), BIG2, BIG1);
  run_gcn(BIG1, 300, 200, P(P_gcn5_W), P(P_gcn5_b), BIG0, g2a, g2b);
  run_gat(BIG0, 200, 100, P(P_gat5_W), P(P_gat5_al), P(P_gat5_ar), P(P_gat5_b), BIG2, BIG1);
  run_gcn(BIG1, 600, 400, P(P_gcn6_W), P(P_gcn6_b), BIG0, g2a, g2b);
  run_gat(BIG0, 400, 800, P(P_gat6_W), P(P_gat6_al), P(P_gat6_ar), P(P_gat6_b), BIG2, BIG1);

  // ---- CNN decoder: BIG1 = (N,48,10,10) -> (N,3,16,16) ----
  {
    float* d0 = BIG2;                          // (N,24,12,12)
    float* d1 = BIG2 + (size_t)NN * 3456;      // (N,12,14,14)
    float* d2 = d1 + (size_t)NN * 2352;        // (N,3,16,16)
    int tot;
    tot = NN * 24 * 12 * 12;
    convt3x3<<<(tot + 255) / 256, 256, 0, stream>>>(BIG1, P(P_dec1_W), P(P_dec1_b), d0,
                                                    48, 24, 10, 10, 1, tot);
    tot = NN * 12 * 14 * 14;
    convt3x3<<<(tot + 255) / 256, 256, 0, stream>>>(d0, P(P_dec2_W), P(P_dec2_b), d1,
                                                    24, 12, 12, 12, 1, tot);
    tot = NN * 3 * 16 * 16;
    convt3x3<<<(tot + 255) / 256, 256, 0, stream>>>(d1, P(P_dec3_W), P(P_dec3_b), d2,
                                                    12, 3, 14, 14, 2, tot);
    // ---- final dense: (N,768) @ (768,36) + b -> d_out ----
    gemm(d2, P(P_dense_W), P(P_dense_b), (float*)d_out, NN, 36, 768);
  }
}
